// TransformerDecoderLayer_17643725652492
// MI455X (gfx1250) — matrix operand order repaired and checked by reading, not compiled, not run on hardware
//
#include <hip/hip_runtime.h>
#include <hip/hip_bf16.h>

// ---------------------------------------------------------------------------
// Types for CDNA5 WMMA (gfx1250, wave32)
// ---------------------------------------------------------------------------
typedef __bf16 bf16;
typedef __attribute__((ext_vector_type(16))) __bf16 bf16x16;
typedef __attribute__((ext_vector_type(8)))  __bf16 bf16x8;
typedef __attribute__((ext_vector_type(8)))  float  f32x8;

union FragAB { bf16x16 v; bf16x8 h[2]; };
union Acc8   { f32x8   v; float  f[8]; };

// A-matrix fragment (16x32 bf16): lane<16 -> K 0..15, lane>=16 -> K 16..31
// (kb=8*(lane>>4), start 2*kb). Two contiguous 16B chunks per lane.
__device__ inline bf16x16 load_frag_a(const bf16* __restrict__ row, int kb) {
  FragAB f;
  f.h[0] = *(const bf16x8*)(row + 2 * kb);
  f.h[1] = *(const bf16x8*)(row + 2 * kb + 8);
  return f.v;
}
// B-matrix fragment (32x16 bf16): lane<16 -> K 0..15, lane>=16 -> K 16..31,
// contiguous along K for the lane's column. ks = 16*(lane>>4).
__device__ inline bf16x16 load_frag_b(const bf16* __restrict__ row, int ks) {
  FragAB f;
  f.h[0] = *(const bf16x8*)(row + ks);
  f.h[1] = *(const bf16x8*)(row + ks + 8);
  return f.v;
}
__device__ inline f32x8 wmma_bf16(bf16x16 a, bf16x16 b, f32x8 c) {
  return __builtin_amdgcn_wmma_f32_16x16x32_bf16(false, a, false, b,
                                                 (short)0, c, false, false);
}

// ---------------------------------------------------------------------------
// CDNA5 async global->LDS copy (ASYNCcnt path), 16B per lane.
// lds_off: byte offset into the wave's LDS aperture (low 32 bits of a
// generic pointer to a __shared__ object). gptr: 64-bit global address.
// ---------------------------------------------------------------------------
__device__ inline void async_ld16(unsigned lds_off, const bf16* gptr) {
  asm volatile("global_load_async_to_lds_b128 %0, %1, off"
               :: "v"(lds_off), "v"(gptr) : "memory");
}
__device__ inline void wait_async0() {
  asm volatile("s_wait_asynccnt 0" ::: "memory");
}
__device__ inline unsigned lds_off_of(const void* p) {
  return (unsigned)(uintptr_t)p;
}

// ---------------------------------------------------------------------------
// Problem constants
// ---------------------------------------------------------------------------
constexpr int cL  = 2048;
constexpr int cS  = 2048;
constexpr int cB  = 4;
constexpr int cE  = 1024;
constexpr int cH  = 16;
constexpr int cHD = 64;
constexpr int cFF = 4096;
constexpr int cT  = cL * cB;   // 8192 token rows

// ---------------------------------------------------------------------------
// f32 -> bf16 conversion (grid-stride)
// ---------------------------------------------------------------------------
__global__ __launch_bounds__(256)
void cvt_f32_bf16_kernel(const float* __restrict__ src, bf16* __restrict__ dst, int n) {
  int i = blockIdx.x * blockDim.x + threadIdx.x;
  int stride = gridDim.x * blockDim.x;
  for (; i < n; i += stride) dst[i] = (bf16)src[i];
}

// ---------------------------------------------------------------------------
// GEMM: C[M,N] = A[M,K](bf16,row-major) @ W[N,K](bf16,row-major)^T + bias[N]
// Block tile 128x64, 8 waves of 32x32, BK=64, double-buffered LDS with
// async global->LDS prefetch of the next K tile. M%128==0, N%64==0, K%64==0.
// ---------------------------------------------------------------------------
template<bool RELU, bool OUTBF>
__global__ __launch_bounds__(256)
void gemm_bf16_kernel(const bf16* __restrict__ A, const bf16* __restrict__ W,
                      const float* __restrict__ bias,
                      float* __restrict__ Cf, bf16* __restrict__ Cb,
                      int M, int N, int K)
{
  constexpr int BM = 128, BN = 64, BK = 64, LDA = BK + 8;   // 72
  constexpr int A_ELEMS = BM * LDA;                          // 9216
  constexpr int B_ELEMS = BN * LDA;                          // 4608
  __shared__ __align__(16) bf16 As[2 * A_ELEMS];
  __shared__ __align__(16) bf16 Bs[2 * B_ELEMS];

  const int tid  = threadIdx.x;
  const int lane = tid & 31;
  const int wave = tid >> 5;
  const int wm   = wave >> 1;        // 0..3
  const int wn   = wave & 1;         // 0..1
  const int m0   = blockIdx.y * BM;
  const int n0   = blockIdx.x * BN;

  const int mm = lane & 15;
  const int hi = lane >> 4;
  const int kb = hi * 8;
  const int ks = hi * 16;

  const unsigned asBase = lds_off_of(As);
  const unsigned bsBase = lds_off_of(Bs);

  // per-thread staging coordinates (fixed trip counts, no exec branching)
  // A tile: 128 rows x 64 elems = 1024 16B-chunks -> 4 per thread
  // W tile:  64 rows x 64 elems =  512 16B-chunks -> 2 per thread
  auto stage = [&](int k0, int buf) {
    unsigned aB = asBase + (unsigned)buf * (A_ELEMS * 2);
    unsigned bB = bsBase + (unsigned)buf * (B_ELEMS * 2);
    #pragma unroll
    for (int i = 0; i < 4; ++i) {
      int c = tid + i * 256;
      int r = c >> 3, kk = (c & 7) * 8;
      async_ld16(aB + (unsigned)(r * LDA + kk) * 2,
                 A + (size_t)(m0 + r) * K + k0 + kk);
    }
    #pragma unroll
    for (int i = 0; i < 2; ++i) {
      int c = tid + i * 256;
      int r = c >> 3, kk = (c & 7) * 8;
      async_ld16(bB + (unsigned)(r * LDA + kk) * 2,
                 W + (size_t)(n0 + r) * K + k0 + kk);
    }
  };

  f32x8 c00{}, c01{}, c10{}, c11{};

  stage(0, 0);
  wait_async0();
  __syncthreads();

  int cur = 0;
  for (int k0 = 0; k0 < K; k0 += BK) {
    if (k0 + BK < K) stage(k0 + BK, cur ^ 1);   // prefetch next tile

    const bf16* aBase = &As[cur * A_ELEMS];
    const bf16* bBase = &Bs[cur * B_ELEMS];
    #pragma unroll
    for (int s = 0; s < 2; ++s) {               // two 32-wide k-steps
      const bf16* arow0 = &aBase[(wm * 32 + mm) * LDA + s * 32];
      const bf16* arow1 = arow0 + 16 * LDA;
      const bf16* brow0 = &bBase[(wn * 32 + mm) * LDA + s * 32];
      const bf16* brow1 = brow0 + 16 * LDA;
      bf16x16 a0 = load_frag_a(arow0, kb);
      bf16x16 a1 = load_frag_a(arow1, kb);
      bf16x16 b0 = load_frag_b(brow0, ks);
      bf16x16 b1 = load_frag_b(brow1, ks);
      c00 = wmma_bf16(a0, b0, c00);
      c01 = wmma_bf16(a0, b1, c01);
      c10 = wmma_bf16(a1, b0, c10);
      c11 = wmma_bf16(a1, b1, c11);
    }

    wait_async0();       // next tile landed
    __syncthreads();     // everyone done reading cur / writing next
    cur ^= 1;
  }

  // epilogue: C element (r of v8f) -> row = base + 8*hi + r, col = lane&15
  const int col0 = n0 + wn * 32 + mm;
  const int col1 = col0 + 16;
  const float bias0 = bias[col0];
  const float bias1 = bias[col1];
  const int rbase = m0 + wm * 32 + hi * 8;

  Acc8 u00, u01, u10, u11;
  u00.v = c00; u01.v = c01; u10.v = c10; u11.v = c11;
  #pragma unroll
  for (int r = 0; r < 8; ++r) {
    int row  = rbase + r;
    int row2 = row + 16;
    float v00 = u00.f[r] + bias0, v01 = u01.f[r] + bias1;
    float v10 = u10.f[r] + bias0, v11 = u11.f[r] + bias1;
    if (RELU) {
      v00 = fmaxf(v00, 0.f); v01 = fmaxf(v01, 0.f);
      v10 = fmaxf(v10, 0.f); v11 = fmaxf(v11, 0.f);
    }
    if (OUTBF) {
      Cb[(size_t)row  * N + col0] = (bf16)v00;
      Cb[(size_t)row  * N + col1] = (bf16)v01;
      Cb[(size_t)row2 * N + col0] = (bf16)v10;
      Cb[(size_t)row2 * N + col1] = (bf16)v11;
    } else {
      Cf[(size_t)row  * N + col0] = v00;
      Cf[(size_t)row  * N + col1] = v01;
      Cf[(size_t)row2 * N + col0] = v10;
      Cf[(size_t)row2 * N + col1] = v11;
    }
  }
}

// ---------------------------------------------------------------------------
// Flash-style attention. Q:(Lq,B,E) bf16, K/V:(Ls,B,E) bf16, per head HD=64.
// Block: 128 thr = 4 waves; block covers 64 query rows of one (b,h).
// Wave handles 16 query rows; keys streamed in 64-wide tiles via async LDS.
// ---------------------------------------------------------------------------
template<bool CAUSAL>
__global__ __launch_bounds__(128)
void attention_kernel(const bf16* __restrict__ Q, const bf16* __restrict__ K,
                      const bf16* __restrict__ V, bf16* __restrict__ Out,
                      int Ls, int Bsz, int nH)
{
  constexpr int LDQ = 72;      // 64 + 8 pad (bf16 elems)
  __shared__ __align__(16) bf16 Qs[64 * LDQ];
  __shared__ __align__(16) bf16 Ks[64 * LDQ];
  __shared__ __align__(16) bf16 Vs[64 * LDQ];
  __shared__ __align__(16) bf16 Ps[4 * 16 * LDQ];   // per-wave 16x64 P tile

  const int tid   = threadIdx.x;
  const int lane  = tid & 31;
  const int wave  = tid >> 5;
  const int qBase = blockIdx.x * 64;
  const int h     = blockIdx.y % nH;
  const int b     = blockIdx.y / nH;
  const int Etot  = nH * cHD;

  const int mm = lane & 15;
  const int hi = lane >> 4;
  const int kb = hi * 8;
  const int ks = hi * 16;

  const unsigned qsBase = lds_off_of(Qs);
  const unsigned ksBase = lds_off_of(Ks);
  const unsigned vsBase = lds_off_of(Vs);

  // stage Q tile (64 rows x 64 hd): 512 chunks -> 4 per thread
  #pragma unroll
  for (int i = 0; i < 4; ++i) {
    int c  = tid + i * 128;
    int r  = c >> 3;
    int kk = (c & 7) * 8;
    async_ld16(qsBase + (unsigned)(r * LDQ + kk) * 2,
               Q + ((size_t)(qBase + r) * Bsz + b) * Etot + h * cHD + kk);
  }
  wait_async0();
  __syncthreads();

  // per-wave Q A-fragments (rows wave*16 .. +15), ksub 0/1 -> hd 0..31 / 32..63
  bf16x16 qf[2];
  const bf16* qrow = &Qs[(wave * 16 + mm) * LDQ];
  qf[0] = load_frag_a(qrow,      kb);
  qf[1] = load_frag_a(qrow + 32, kb);

  float m8[8], l8[8];
  Acc8 o[4];
  #pragma unroll
  for (int r = 0; r < 8; ++r) { m8[r] = -1e30f; l8[r] = 0.f; }
  #pragma unroll
  for (int t = 0; t < 4; ++t) o[t].v = f32x8{};

  const float scale = 0.125f;   // 1/sqrt(64)
  const int sEnd = CAUSAL ? ((qBase + 64) < Ls ? (qBase + 64) : Ls) : Ls;

  bf16* prow = &Ps[wave * 16 * LDQ];

  for (int sBase = 0; sBase < sEnd; sBase += 64) {
    __syncthreads();           // prior reads of Ks/Vs finished everywhere
    #pragma unroll
    for (int i = 0; i < 4; ++i) {
      int c  = tid + i * 128;
      int r  = c >> 3;
      int kk = (c & 7) * 8;
      size_t src = ((size_t)(sBase + r) * Bsz + b) * Etot + h * cHD + kk;
      unsigned dst = (unsigned)(r * LDQ + kk) * 2;
      async_ld16(ksBase + dst, K + src);
      async_ld16(vsBase + dst, V + src);
    }
    wait_async0();
    __syncthreads();

    // scores: 4 x (16x16) tiles, kdim = 64 over head-dim
    Acc8 sc[4];
    #pragma unroll
    for (int t = 0; t < 4; ++t) {
      f32x8 acc{};
      const bf16* krow = &Ks[(t * 16 + mm) * LDQ];
      acc = wmma_bf16(qf[0], load_frag_b(krow,      ks), acc);
      acc = wmma_bf16(qf[1], load_frag_b(krow + 32, ks), acc);
      sc[t].v = acc;
    }

    // scale + causal mask + row max
    float rmax[8];
    #pragma unroll
    for (int r = 0; r < 8; ++r) rmax[r] = -1e30f;
    #pragma unroll
    for (int t = 0; t < 4; ++t) {
      int col = sBase + t * 16 + mm;
      #pragma unroll
      for (int r = 0; r < 8; ++r) {
        float s = sc[t].f[r] * scale;
        if (CAUSAL) {
          int rowq = qBase + wave * 16 + hi * 8 + r;
          if (col > rowq) s = -1e30f;
        }
        sc[t].f[r] = s;
        rmax[r] = fmaxf(rmax[r], s);
      }
    }
    #pragma unroll
    for (int msk = 1; msk <= 8; msk <<= 1) {
      #pragma unroll
      for (int r = 0; r < 8; ++r)
        rmax[r] = fmaxf(rmax[r], __shfl_xor(rmax[r], msk, 32));
    }

    // online softmax update, write P tile (bf16) to wave-private LDS
    float alpha[8], rsum[8];
    #pragma unroll
    for (int r = 0; r < 8; ++r) {
      float mn = fmaxf(m8[r], rmax[r]);
      alpha[r] = __expf(m8[r] - mn);
      m8[r] = mn;
      rsum[r] = 0.f;
    }
    #pragma unroll
    for (int t = 0; t < 4; ++t) {
      #pragma unroll
      for (int r = 0; r < 8; ++r) {
        float p = __expf(sc[t].f[r] - m8[r]);
        rsum[r] += p;
        prow[(hi * 8 + r) * LDQ + t * 16 + mm] = (bf16)p;
      }
    }
    #pragma unroll
    for (int msk = 1; msk <= 8; msk <<= 1) {
      #pragma unroll
      for (int r = 0; r < 8; ++r)
        rsum[r] += __shfl_xor(rsum[r], msk, 32);
    }
    #pragma unroll
    for (int r = 0; r < 8; ++r) l8[r] = l8[r] * alpha[r] + rsum[r];
    #pragma unroll
    for (int t = 0; t < 4; ++t)
      #pragma unroll
      for (int r = 0; r < 8; ++r) o[t].f[r] *= alpha[r];

    // O += P @ V  (kdim = 64 keys; V columns gathered from LDS)
    #pragma unroll
    for (int s2 = 0; s2 < 2; ++s2) {
      bf16x16 pa = load_frag_a(&prow[mm * LDQ] + s2 * 32, kb);
      #pragma unroll
      for (int t = 0; t < 4; ++t) {
        FragAB vb;
        #pragma unroll
        for (int j = 0; j < 8; ++j) {
          vb.h[0][j] = Vs[(s2 * 32 + ks + j)     * LDQ + t * 16 + mm];
          vb.h[1][j] = Vs[(s2 * 32 + ks + 8 + j) * LDQ + t * 16 + mm];
        }
        o[t].v = wmma_bf16(pa, vb.v, o[t].v);
      }
    }
  }

  float inv[8];
  #pragma unroll
  for (int r = 0; r < 8; ++r) inv[r] = 1.0f / l8[r];
  #pragma unroll
  for (int t = 0; t < 4; ++t) {
    #pragma unroll
    for (int r = 0; r < 8; ++r) {
      int l = qBase + wave * 16 + hi * 8 + r;
      Out[((size_t)l * Bsz + b) * Etot + h * cHD + t * 16 + mm] =
          (bf16)(o[t].f[r] * inv[r]);
    }
  }
}

// ---------------------------------------------------------------------------
// residual add + LayerNorm over E=1024, one block per token row
// ---------------------------------------------------------------------------
__device__ inline float block_sum_1024(float v) {
  __shared__ float red[8];
  int lane = threadIdx.x & 31;
  int w    = threadIdx.x >> 5;
  #pragma unroll
  for (int m = 16; m >= 1; m >>= 1) v += __shfl_xor(v, m, 32);
  __syncthreads();
  if (lane == 0) red[w] = v;
  __syncthreads();
  float s = 0.f;
  #pragma unroll
  for (int i = 0; i < 8; ++i) s += red[i];
  return s;
}

__global__ __launch_bounds__(256)
void add_ln_kernel(const float* __restrict__ A, const float* __restrict__ R,
                   const float* __restrict__ g, const float* __restrict__ be,
                   float* __restrict__ outF, bf16* __restrict__ outB)
{
  size_t base = (size_t)blockIdx.x * cE;
  float v[4];
  float s = 0.f;
  #pragma unroll
  for (int i = 0; i < 4; ++i) {
    int idx = threadIdx.x + i * 256;
    v[i] = A[base + idx] + R[base + idx];
    s += v[i];
  }
  float mean = block_sum_1024(s) * (1.0f / 1024.0f);
  float s2 = 0.f;
  #pragma unroll
  for (int i = 0; i < 4; ++i) { float d = v[i] - mean; s2 += d * d; }
  float var  = block_sum_1024(s2) * (1.0f / 1024.0f);
  float rstd = rsqrtf(var + 1e-5f);
  #pragma unroll
  for (int i = 0; i < 4; ++i) {
    int idx = threadIdx.x + i * 256;
    float o = (v[i] - mean) * rstd * g[idx] + be[idx];
    outF[base + idx] = o;
    if (outB) outB[base + idx] = (bf16)o;
  }
}

// ---------------------------------------------------------------------------
// launch
// ---------------------------------------------------------------------------
extern "C" void kernel_launch(void* const* d_in, const int* in_sizes, int n_in,
                              void* d_out, int out_size, void* d_ws, size_t ws_size,
                              hipStream_t stream) {
  (void)in_sizes; (void)n_in; (void)out_size; (void)ws_size;

  // setup_inputs() dict order
  const float* tgt    = (const float*)d_in[0];
  const float* memory = (const float*)d_in[1];
  // d_in[2] tgt_mask: upper-triangular -inf, implemented analytically
  const float* sa_Wq = (const float*)d_in[3];  const float* sa_bq = (const float*)d_in[4];
  const float* sa_Wk = (const float*)d_in[5];  const float* sa_bk = (const float*)d_in[6];
  const float* sa_Wv = (const float*)d_in[7];  const float* sa_bv = (const float*)d_in[8];
  const float* sa_Wo = (const float*)d_in[9];  const float* sa_bo = (const float*)d_in[10];
  const float* ca_Wq = (const float*)d_in[11]; const float* ca_bq = (const float*)d_in[12];
  const float* ca_Wk = (const float*)d_in[13]; const float* ca_bk = (const float*)d_in[14];
  const float* ca_Wv = (const float*)d_in[15]; const float* ca_bv = (const float*)d_in[16];
  const float* ca_Wo = (const float*)d_in[17]; const float* ca_bo = (const float*)d_in[18];
  const float* W1 = (const float*)d_in[19];    const float* b1 = (const float*)d_in[20];
  const float* W2 = (const float*)d_in[21];    const float* b2 = (const float*)d_in[22];
  const float* ln1_g = (const float*)d_in[23]; const float* ln1_b = (const float*)d_in[24];
  const float* ln2_g = (const float*)d_in[25]; const float* ln2_b = (const float*)d_in[26];
  const float* ln3_g = (const float*)d_in[27]; const float* ln3_b = (const float*)d_in[28];
  float* out = (float*)d_out;

  char* wsp = (char*)d_ws;
  size_t off = 0;
  auto alloc = [&](size_t bytes) -> void* {
    void* p = wsp + off;
    off += (bytes + 255) & ~(size_t)255;
    return p;
  };

  const size_t EE = (size_t)cE * cE;
  bf16* w_saWq = (bf16*)alloc(EE * 2);
  bf16* w_saWk = (bf16*)alloc(EE * 2);
  bf16* w_saWv = (bf16*)alloc(EE * 2);
  bf16* w_saWo = (bf16*)alloc(EE * 2);
  bf16* w_caWq = (bf16*)alloc(EE * 2);
  bf16* w_caWk = (bf16*)alloc(EE * 2);
  bf16* w_caWv = (bf16*)alloc(EE * 2);
  bf16* w_caWo = (bf16*)alloc(EE * 2);
  bf16* w_W1   = (bf16*)alloc((size_t)cFF * cE * 2);
  bf16* w_W2   = (bf16*)alloc((size_t)cE * cFF * 2);
  bf16* x_bf   = (bf16*)alloc((size_t)cT * cE * 2);   // tgt bf16, then x1/x2 bf16
  bf16* mem_bf = (bf16*)alloc((size_t)cS * cB * cE * 2);
  bf16* q_bf   = (bf16*)alloc((size_t)cT * cE * 2);
  bf16* k_bf   = (bf16*)alloc((size_t)cT * cE * 2);
  bf16* v_bf   = (bf16*)alloc((size_t)cT * cE * 2);
  bf16* at_bf  = (bf16*)alloc((size_t)cT * cE * 2);
  float* proj_f = (float*)alloc((size_t)cT * cE * 4);
  float* x_f    = (float*)alloc((size_t)cT * cE * 4);
  bf16* ffn_bf  = (bf16*)alloc((size_t)cT * cFF * 2);

  auto cvt = [&](const float* s, bf16* d, int n) {
    cvt_f32_bf16_kernel<<<4096, 256, 0, stream>>>(s, d, n);
  };
  cvt(sa_Wq, w_saWq, (int)EE);  cvt(sa_Wk, w_saWk, (int)EE);
  cvt(sa_Wv, w_saWv, (int)EE);  cvt(sa_Wo, w_saWo, (int)EE);
  cvt(ca_Wq, w_caWq, (int)EE);  cvt(ca_Wk, w_caWk, (int)EE);
  cvt(ca_Wv, w_caWv, (int)EE);  cvt(ca_Wo, w_caWo, (int)EE);
  cvt(W1, w_W1, cFF * cE);
  cvt(W2, w_W2, cE * cFF);
  cvt(tgt, x_bf, cT * cE);
  cvt(memory, mem_bf, cS * cB * cE);

  auto ggrid = [](int M, int N) { return dim3((unsigned)(N / 64), (unsigned)(M / 128)); };

  // ---- self-attention block ----
  gemm_bf16_kernel<false, true><<<ggrid(cT, cE), 256, 0, stream>>>(
      x_bf, w_saWq, sa_bq, nullptr, q_bf, cT, cE, cE);
  gemm_bf16_kernel<false, true><<<ggrid(cT, cE), 256, 0, stream>>>(
      x_bf, w_saWk, sa_bk, nullptr, k_bf, cT, cE, cE);
  gemm_bf16_kernel<false, true><<<ggrid(cT, cE), 256, 0, stream>>>(
      x_bf, w_saWv, sa_bv, nullptr, v_bf, cT, cE, cE);
  attention_kernel<true><<<dim3(cL / 64, cB * cH), 128, 0, stream>>>(
      q_bf, k_bf, v_bf, at_bf, cL, cB, cH);
  gemm_bf16_kernel<false, false><<<ggrid(cT, cE), 256, 0, stream>>>(
      at_bf, w_saWo, sa_bo, proj_f, nullptr, cT, cE, cE);
  add_ln_kernel<<<cT, 256, 0, stream>>>(tgt, proj_f, ln1_g, ln1_b, x_f, x_bf);

  // ---- cross-attention block ----
  gemm_bf16_kernel<false, true><<<ggrid(cT, cE), 256, 0, stream>>>(
      x_bf, w_caWq, ca_bq, nullptr, q_bf, cT, cE, cE);
  gemm_bf16_kernel<false, true><<<ggrid(cT, cE), 256, 0, stream>>>(
      mem_bf, w_caWk, ca_bk, nullptr, k_bf, cT, cE, cE);
  gemm_bf16_kernel<false, true><<<ggrid(cT, cE), 256, 0, stream>>>(
      mem_bf, w_caWv, ca_bv, nullptr, v_bf, cT, cE, cE);
  attention_kernel<false><<<dim3(cL / 64, cB * cH), 128, 0, stream>>>(
      q_bf, k_bf, v_bf, at_bf, cS, cB, cH);
  gemm_bf16_kernel<false, false><<<ggrid(cT, cE), 256, 0, stream>>>(
      at_bf, w_caWo, ca_bo, proj_f, nullptr, cT, cE, cE);
  add_ln_kernel<<<cT, 256, 0, stream>>>(x_f, proj_f, ln2_g, ln2_b, x_f, x_bf);

  // ---- FFN block ----
  gemm_bf16_kernel<true, true><<<ggrid(cT, cFF), 256, 0, stream>>>(
      x_bf, w_W1, b1, nullptr, ffn_bf, cT, cFF, cE);
  gemm_bf16_kernel<false, false><<<ggrid(cT, cE), 256, 0, stream>>>(
      ffn_bf, w_W2, b2, proj_f, nullptr, cT, cE, cFF);
  add_ln_kernel<<<cT, 256, 0, stream>>>(x_f, proj_f, ln3_g, ln3_b, out, nullptr);
}
